// CompressedLinear_7919919694465
// MI455X (gfx1250) — compile-verified, matrix operand order checked by reading, not run on hardware
//
#include <hip/hip_runtime.h>
#include <stdint.h>

typedef __attribute__((ext_vector_type(16))) _Float16 v16h;
typedef __attribute__((ext_vector_type(8)))  _Float16 v8h;
typedef __attribute__((ext_vector_type(4)))  _Float16 v4h;
typedef __attribute__((ext_vector_type(8)))  float    v8f;
typedef __attribute__((ext_vector_type(4)))  float    v4f;
typedef __attribute__((ext_vector_type(4)))  int      v4i;

#define IN_F   4096
#define OUT_F  4096
#define M_TOT  8192           // 4 * 2048
#define SEQ    4
#define NCODE_STREAM (OUT_F * IN_F / SEQ)   // 4,194,304 codewords

#define BM  128
#define BN  256
#define BK  32
#define BKP 40                 // padded LDS row stride in halves (20 banks -> conflict-free)

// low 32 bits of a generic pointer to __shared__ == wave-relative LDS byte offset
__device__ __forceinline__ uint32_t lds_off(const void* p) {
    return (uint32_t)(uintptr_t)p;
}

// ---------------------------------------------------------------------------
// Kernel 1: decompress codeword stream -> f16 weight matrix W[OUT_F][IN_F]
// ---------------------------------------------------------------------------
__global__ __launch_bounds__(256)
void decomp_kernel(const int* __restrict__ codes,
                   const int* __restrict__ table,
                   _Float16* __restrict__ w) {
    const int i = blockIdx.x * blockDim.x + threadIdx.x;   // 0 .. NCODE_STREAM-1
    const int code = codes[i];
    const v4i t = *(const v4i*)(table + (size_t)code * 4); // 16B aligned gather
    v4h o;
    #pragma unroll
    for (int j = 0; j < 4; ++j) o[j] = (_Float16)t[j];     // ints 0..255 exact in f16
    *(v4h*)(w + (size_t)i * 4) = o;
}

// ---------------------------------------------------------------------------
// Kernel 2: out = x @ W^T + bias via v_wmma_f32_16x16x32_f16
//   block tile 128(M) x 256(N), BK=32; 8 waves in 2(M) x 4(N); wave tile 64x64
//   A: global f32 -> regs -> cvt f16 -> LDS (double buffered)
//   B: global f16 -> LDS via global_load_async_to_lds_b128 (ASYNCcnt)
// ---------------------------------------------------------------------------
__global__ __launch_bounds__(256)
void gemm_kernel(const float* __restrict__ x,        // [M_TOT][IN_F] f32
                 const _Float16* __restrict__ w,     // [OUT_F][IN_F] f16
                 const float* __restrict__ bias,     // [OUT_F]
                 float* __restrict__ out) {          // [M_TOT][OUT_F] f32
    __shared__ _Float16 As[2][BM * BKP];   // 2 x 10240 B
    __shared__ _Float16 Bs[2][BN * BKP];   // 2 x 20480 B

    const int tid = threadIdx.x;
    const int m0  = blockIdx.y * BM;
    const int n0  = blockIdx.x * BN;

    // ---- A staging map: each thread owns half a tile row (16 f32 along K)
    const int arow = tid >> 1;             // 0..127
    const int acol = (tid & 1) * 16;       // 0 or 16
    const float* xg = x + (size_t)(m0 + arow) * IN_F + acol;

    // ---- B async staging map: each thread owns one full 64B tile row slab
    const _Float16* wg = w + (size_t)(n0 + tid) * IN_F;   // tid = 0..255 -> B row
    const uint32_t ldsB0 = lds_off(&Bs[0][tid * BKP]);
    const uint32_t ldsB1 = lds_off(&Bs[1][tid * BKP]);

    // ---- wave / lane map
    const int lane  = tid & 31;
    const int wave  = tid >> 5;
    const int wm    = (wave & 1) * 64;     // 2 waves along M
    const int wn    = (wave >> 1) * 64;    // 4 waves along N
    const int lr    = lane & 15;           // fragment row/col index
    const int khalf = lane >> 4;           // ISA 16-bit operand K-chunk select
    const int klo   = khalf * 8;

    v8f c[4][4];
    #pragma unroll
    for (int i = 0; i < 4; ++i)
        #pragma unroll
        for (int j = 0; j < 4; ++j) {
            v8f z = {};
            c[i][j] = z;
        }

    v4f ar[4];   // A global->reg prefetch (16 f32)

    auto loadA_global = [&](int k) {
        #pragma unroll
        for (int q = 0; q < 4; ++q) ar[q] = *(const v4f*)(xg + k + q * 4);
    };
    auto storeA_lds = [&](int buf) {
        _Float16* ap = &As[buf][arow * BKP + acol];
        v8h h0, h1;
        #pragma unroll
        for (int j = 0; j < 4; ++j) { h0[j]     = (_Float16)ar[0][j];
                                      h0[j + 4] = (_Float16)ar[1][j];
                                      h1[j]     = (_Float16)ar[2][j];
                                      h1[j + 4] = (_Float16)ar[3][j]; }
        *(v8h*)(ap)     = h0;
        *(v8h*)(ap + 8) = h1;
    };
    // B tile row (64B) DMA'd straight into LDS; same inst_offset applies to
    // both global and LDS sides (ISA async pseudocode), rows are contiguous on both.
    auto issue_asyncB = [&](int k, int buf) {
        const void*    g = (const void*)(wg + k);
        const uint32_t l = buf ? ldsB1 : ldsB0;
        asm volatile(
            "global_load_async_to_lds_b128 %0, %1, off\n\t"
            "global_load_async_to_lds_b128 %0, %1, off offset:16\n\t"
            "global_load_async_to_lds_b128 %0, %1, off offset:32\n\t"
            "global_load_async_to_lds_b128 %0, %1, off offset:48"
            :: "v"(l), "v"(g) : "memory");
    };
    // ISA 16-bit operand layout: lane<16 holds K {0-7,16-23}; lane>=16 holds {8-15,24-31}
    auto load_frag = [&](const _Float16* base, int row) -> v16h {
        const _Float16* p = base + row * BKP + klo;
        const v8h c0 = *(const v8h*)(p);
        const v8h c1 = *(const v8h*)(p + 16);
        v16h r;
        #pragma unroll
        for (int j = 0; j < 8; ++j) { r[j] = c0[j]; r[j + 8] = c1[j]; }
        return r;
    };

    // ---- prologue: fill buffer 0
    issue_asyncB(0, 0);
    loadA_global(0);
    storeA_lds(0);
    asm volatile("s_wait_asynccnt 0" ::: "memory");
    __syncthreads();

    constexpr int KSTEPS = IN_F / BK;   // 128
    for (int kt = 0; kt < KSTEPS; ++kt) {
        const int cur = kt & 1;
        const int nxt = cur ^ 1;
        const bool more = (kt + 1 < KSTEPS);

        if (more) {
            issue_asyncB((kt + 1) * BK, nxt);   // ASYNCcnt-tracked DMA, overlaps WMMA
            loadA_global((kt + 1) * BK);        // LOADcnt-tracked, overlaps WMMA
        }

        const _Float16* Ac = &As[cur][0];
        const _Float16* Bc = &Bs[cur][0];
        v16h a[4], b[4];
        #pragma unroll
        for (int i = 0; i < 4; ++i) a[i] = load_frag(Ac, wm + i * 16 + lr);
        #pragma unroll
        for (int j = 0; j < 4; ++j) b[j] = load_frag(Bc, wn + j * 16 + lr);

        #pragma unroll
        for (int i = 0; i < 4; ++i)
            #pragma unroll
            for (int j = 0; j < 4; ++j)
                c[i][j] = __builtin_amdgcn_wmma_f32_16x16x32_f16(
                    /*neg_a=*/false, a[i], /*neg_b=*/false, b[j],
                    /*c_mod=*/(short)0, c[i][j],
                    /*reuse_a=*/false, /*reuse_b=*/false);

        if (more) storeA_lds(nxt);              // writes idle buffer: no extra barrier
        asm volatile("s_wait_asynccnt 0" ::: "memory");
        __syncthreads();                        // single barrier per K-step
    }

    // ---- epilogue: C/D layout -> VGPR v holds M = v + 8*khalf, N = lr; fuse bias
    #pragma unroll
    for (int j = 0; j < 4; ++j) {
        const int n   = n0 + wn + j * 16 + lr;
        const float bv = bias[n];
        #pragma unroll
        for (int i = 0; i < 4; ++i) {
            const int mbase = m0 + wm + i * 16 + khalf * 8;
            float* op = out + (size_t)mbase * OUT_F + n;
            #pragma unroll
            for (int v = 0; v < 8; ++v)
                op[(size_t)v * OUT_F] = c[i][j][v] + bv;
        }
    }
}

// ---------------------------------------------------------------------------
extern "C" void kernel_launch(void* const* d_in, const int* in_sizes, int n_in,
                              void* d_out, int out_size, void* d_ws, size_t ws_size,
                              hipStream_t stream) {
    const float* x     = (const float*)d_in[0];
    const int*   table = (const int*)  d_in[1];
    const int*   codes = (const int*)  d_in[2];
    const float* bias  = (const float*)d_in[3];
    float*       out   = (float*)d_out;

    _Float16* w16 = (_Float16*)d_ws;   // needs OUT_F*IN_F*2 = 32 MiB of scratch

    // 1) decompress codeword stream into f16 weights (L2-resident thereafter)
    decomp_kernel<<<NCODE_STREAM / 256, 256, 0, stream>>>(codes, table, w16);

    // 2) WMMA GEMM: grid = (N/BN, M/BM)
    dim3 grid(OUT_F / BN, M_TOT / BM);
    gemm_kernel<<<grid, 256, 0, stream>>>(x, w16, bias, out);
}
